// MEGNetBlock_13752485282409
// MI455X (gfx1250) — compile-verified, matrix operand order checked by reading, not run on hardware
//
#include <hip/hip_runtime.h>

typedef __attribute__((ext_vector_type(16))) _Float16 v16h;
typedef __attribute__((ext_vector_type(8)))  _Float16 v8h;
typedef __attribute__((ext_vector_type(4)))  _Float16 h4;
typedef __attribute__((ext_vector_type(8)))  float    v8f;

#define N_NODES 50000
#define N_EDGES 800000
#define D 64

// Workspace layout (float offsets)
#define OFF_V      0                          // v encoding, N_NODES*D
#define OFF_VESUM  (N_NODES * D)              // segment sum of e_new by dst
#define OFF_CNT    (OFF_VESUM + N_NODES * D)  // segment counts
#define OFF_ESUM   (OFF_CNT + N_NODES)        // column sums of e_new (for mean)
#define OFF_VSUM   (OFF_ESUM + D)             // column sums of v_new (for mean)
#define OFF_UENC   (OFF_VSUM + D)             // u encoding (64)
#define ZERO_N     (OFF_UENC - OFF_VESUM)     // floats to zero each launch

// single v_max_num_f32 (NaN -> 0 matches max_num semantics)
__device__ __forceinline__ float relu(float x) { return x > 0.0f ? x : 0.0f; }

// ---- gfx1250 async memory->LDS (ASYNCcnt-tracked DMA) ----
__device__ __forceinline__ void async_lds_load_f32(float* lds_dst, const float* gsrc) {
    unsigned int       laddr = (unsigned int)(unsigned long long)lds_dst; // LDS aperture: addr[31:0]
    unsigned long long gaddr = (unsigned long long)gsrc;
    asm volatile("global_load_async_to_lds_b32 %0, %1, off" :: "v"(laddr), "v"(gaddr) : "memory");
}
__device__ __forceinline__ void wait_asynccnt0() {
    asm volatile("s_wait_asynccnt 0" ::: "memory");
}

// ---- WMMA fragment helpers (wave32, 16x16x32 f16 -> f32) ----
// A 16x32 f16: lane m = lane&15, kb = (lane&16)?8:0; element i:
//   i=0..7  -> K = kb + i          (contiguous 8 halves)
//   i=8..15 -> K = kb + 16 + (i-8) (contiguous 8 halves)
__device__ __forceinline__ int wmma_k_of(int i, int lane) {
    int kb = (lane & 16) ? 8 : 0;
    return (i < 8) ? (kb + i) : (kb + 16 + (i - 8));
}

// A fragment = two ds_load_b128 per lane (16B-aligned chunks)
__device__ __forceinline__ v16h frag_a_lds(const _Float16* smA, int stride, int k0) {
    int lane = threadIdx.x & 31;
    int m = lane & 15;
    int kb = (lane & 16) ? 8 : 0;
    const _Float16* p = smA + m * stride + k0 + kb;
    v8h lo = *(const v8h*)(p);
    v8h hi = *(const v8h*)(p + 16);
    v16h a;
#pragma unroll
    for (int i = 0; i < 8; ++i) { a[i] = lo[i]; a[8 + i] = hi[i]; }
    return a;
}

// B fragments stored fragment-major in LDS: [(kt*4+nt)*32 + lane][16 halves]
__device__ __forceinline__ v16h frag_b_sw(const _Float16* s, int kt, int nt) {
    int lane = threadIdx.x & 31;
    return *(const v16h*)(s + (((kt * 4 + nt) * 32 + lane) << 4));
}

// Convert KxD row-major f32 weights -> fragment-major f16 in LDS
__device__ __forceinline__ void cvt_w_fragB(const float* g, _Float16* s, int K, int tid, int nthr) {
    int total = K * D;
    for (int idx = tid; idx < total; idx += nthr) {
        int i    = idx & 15;
        int lane = (idx >> 4) & 31;
        int nt   = (idx >> 9) & 3;
        int kt   = idx >> 11;
        int k = kt * 32 + wmma_k_of(i, lane);
        int n = nt * 16 + (lane & 15);
        s[idx] = (_Float16)g[k * D + n];
    }
}

__device__ __forceinline__ h4 f4_to_h4(float4 v) {
    h4 r;
    r[0] = (_Float16)v.x; r[1] = (_Float16)v.y; r[2] = (_Float16)v.z; r[3] = (_Float16)v.w;
    return r;
}

// -------------------- zero accumulators --------------------
__global__ void zero_kernel(float* ws) {
    size_t i = (size_t)blockIdx.x * blockDim.x + threadIdx.x;
    if (i < (size_t)ZERO_N) ws[OFF_VESUM + i] = 0.0f;
}

// -------------------- u = relu(state_attr @ duW + dub) --------------------
__global__ __launch_bounds__(64) void uenc_kernel(const float* state_attr, const float* duW,
                                                  const float* dub, float* ws) {
    __shared__ float sa[D];
    int t = threadIdx.x;
    if (t < D) sa[t] = state_attr[t];
    __syncthreads();
    if (t < D) {
        float acc = dub[t];
        for (int k = 0; k < D; ++k) acc += sa[k] * duW[k * D + t];
        ws[OFF_UENC + t] = relu(acc);
    }
}

// -------------------- v = relu(node_feat @ dvW + dvb) --------------------
__global__ __launch_bounds__(128) void venc_kernel(const float* __restrict__ node_feat,
                                                   const float* dvW, const float* dvb, float* ws) {
    __shared__ __align__(32) _Float16 Wv[D * D];
    __shared__ float    bv[D];
    __shared__ __align__(32) _Float16 stg[4 * 16 * D];   // per-wave 16 x 64 f16 (A)
    __shared__ __align__(16) float    ostg[4 * 16 * D];  // per-wave 16 x 64 f32 (out)
    int tid = threadIdx.x, nthr = blockDim.x;
    cvt_w_fragB(dvW, Wv, D, tid, nthr);
    for (int i = tid; i < D; i += nthr) bv[i] = dvb[i];
    __syncthreads();

    int w = tid >> 5, lane = tid & 31;
    int base = blockIdx.x * 64 + w * 16;
    _Float16* A = &stg[w * 16 * D];
    float*    O = &ostg[w * 16 * D];
    for (int t = lane; t < 16 * 16; t += 32) {
        int r = t >> 4, ch = (t & 15) * 4;
        int gr = base + r;
        float4 val = make_float4(0.f, 0.f, 0.f, 0.f);
        if (gr < N_NODES) val = *(const float4*)(node_feat + (size_t)gr * D + ch);
        *(h4*)(A + r * D + ch) = f4_to_h4(val);
    }
    v16h a0 = frag_a_lds(A, D, 0);
    v16h a1 = frag_a_lds(A, D, 32);
#pragma unroll
    for (int nt = 0; nt < 4; ++nt) {
        v8f c;
        float bb = bv[nt * 16 + (lane & 15)];
#pragma unroll
        for (int r = 0; r < 8; ++r) c[r] = bb;
        c = __builtin_amdgcn_wmma_f32_16x16x32_f16(false, a0, false, frag_b_sw(Wv, 0, nt), (short)0, c, false, false);
        c = __builtin_amdgcn_wmma_f32_16x16x32_f16(false, a1, false, frag_b_sw(Wv, 1, nt), (short)0, c, false, false);
#pragma unroll
        for (int r = 0; r < 8; ++r) {
            int row = (lane & 16) ? r + 8 : r;
            O[row * D + nt * 16 + (lane & 15)] = relu(c[r]);
        }
    }
    asm volatile("" ::: "memory");
    float* vout = ws + OFF_V;
    for (int t = lane; t < 16 * 16; t += 32) {
        int r = t >> 4, ch = (t & 15) * 4;
        int gr = base + r;
        if (gr < N_NODES)
            *(float4*)(vout + (size_t)gr * D + ch) = *(const float4*)(O + r * D + ch);
    }
}

// -------------------- fused edge pipeline --------------------
__global__ __launch_bounds__(128) void edge_kernel(const int* __restrict__ src, const int* __restrict__ dst,
                                                   const float* __restrict__ edge_feat,
                                                   const float* deW, const float* deb,
                                                   const float* ceW0, const float* ceb0,
                                                   const float* ceW1, const float* ceb1,
                                                   float* ws, float* __restrict__ out_e) {
    __shared__ __align__(32) _Float16 Wde[D * D];
    __shared__ __align__(32) _Float16 Wc0[4 * D * D];
    __shared__ __align__(32) _Float16 Wc1[D * D];
    __shared__ float    bde[D], bc0[D], bc1[D];
    __shared__ __align__(32) _Float16 u16[D];
    __shared__ float    uf[D];
    __shared__ __align__(32) _Float16 stg[4 * 16 * 4 * D]; // per wave: 16 x 256 f16 e_in
    __shared__ float esum_s[D];

    int tid = threadIdx.x, nthr = blockDim.x;
    const float* uenc = ws + OFF_UENC;
    // async DMA of the u encoding into LDS; overlaps with weight conversion below
    if (tid < D) async_lds_load_f32(&uf[tid], uenc + tid);

    cvt_w_fragB(deW, Wde, D, tid, nthr);
    cvt_w_fragB(ceW0, Wc0, 4 * D, tid, nthr);
    cvt_w_fragB(ceW1, Wc1, D, tid, nthr);
    wait_asynccnt0();
    for (int i = tid; i < D; i += nthr) {
        bde[i] = deb[i]; bc0[i] = ceb0[i]; bc1[i] = ceb1[i];
        u16[i] = (_Float16)uf[i];
        esum_s[i] = 0.0f;
    }
    __syncthreads();

    int w = tid >> 5, lane = tid & 31;
    int base = blockIdx.x * 64 + w * 16;          // N_EDGES % 64 == 0, no tail
    _Float16* EIN = &stg[w * 16 * 256];
    const float* venc = ws + OFF_V;

    // prefetch next block's edge rows (gfx1250 global_prefetch)
    if (tid < 64 && blockIdx.x + 1 < gridDim.x)
        __builtin_prefetch(edge_feat + ((size_t)(blockIdx.x + 1) * 64 + tid) * D, 0, 1);

    // Stage e_in: [v[src] | v[dst] | edge_feat(->e) | u], float4 gathers -> h4 LDS stores
    for (int t = lane; t < 16 * 16; t += 32) {
        int r = t >> 4, ch = (t & 15) * 4;
        int gr = base + r;
        int s = src[gr], d2 = dst[gr];
        *(h4*)(EIN + r * 256 + ch)       = f4_to_h4(*(const float4*)(venc + (size_t)s  * D + ch));
        *(h4*)(EIN + r * 256 + 64 + ch)  = f4_to_h4(*(const float4*)(venc + (size_t)d2 * D + ch));
        *(h4*)(EIN + r * 256 + 128 + ch) = f4_to_h4(*(const float4*)(edge_feat + (size_t)gr * D + ch));
        *(h4*)(EIN + r * 256 + 192 + ch) = *(const h4*)(u16 + ch);
    }

    // e = relu(edge_feat @ deW + deb), overwrite cols 128..191 in-place
    {
        v16h a0 = frag_a_lds(EIN + 128, 256, 0);
        v16h a1 = frag_a_lds(EIN + 128, 256, 32);
        v8f cc[4];
#pragma unroll
        for (int nt = 0; nt < 4; ++nt) {
            float bb = bde[nt * 16 + (lane & 15)];
#pragma unroll
            for (int r = 0; r < 8; ++r) cc[nt][r] = bb;
            cc[nt] = __builtin_amdgcn_wmma_f32_16x16x32_f16(false, a0, false, frag_b_sw(Wde, 0, nt), (short)0, cc[nt], false, false);
            cc[nt] = __builtin_amdgcn_wmma_f32_16x16x32_f16(false, a1, false, frag_b_sw(Wde, 1, nt), (short)0, cc[nt], false, false);
        }
        asm volatile("" ::: "memory");
#pragma unroll
        for (int nt = 0; nt < 4; ++nt)
#pragma unroll
            for (int r = 0; r < 8; ++r) {
                int row = (lane & 16) ? r + 8 : r;
                EIN[row * 256 + 128 + nt * 16 + (lane & 15)] = (_Float16)relu(cc[nt][r]);
            }
    }

    // Layer 1: 256 -> 64, keep all 4 N-tiles in registers
    v8f h[4];
#pragma unroll
    for (int nt = 0; nt < 4; ++nt) {
        float bb = bc0[nt * 16 + (lane & 15)];
#pragma unroll
        for (int r = 0; r < 8; ++r) h[nt][r] = bb;
#pragma unroll
        for (int kt = 0; kt < 8; ++kt) {
            v16h a = frag_a_lds(EIN, 256, kt * 32);
            h[nt] = __builtin_amdgcn_wmma_f32_16x16x32_f16(false, a, false, frag_b_sw(Wc0, kt, nt), (short)0, h[nt], false, false);
        }
    }
    asm volatile("" ::: "memory");
    // relu(h) -> EIN cols 0..63 (v[src] no longer needed)
#pragma unroll
    for (int nt = 0; nt < 4; ++nt)
#pragma unroll
        for (int r = 0; r < 8; ++r) {
            int row = (lane & 16) ? r + 8 : r;
            EIN[row * 256 + nt * 16 + (lane & 15)] = (_Float16)relu(h[nt][r]);
        }

    // Layer 2: 64 -> 64; results -> f32 LDS tile (reuse EIN), then coalesced epilogue
    float* vesum = ws + OFF_VESUM;
    float* cntp  = ws + OFF_CNT;
    float* esum  = ws + OFF_ESUM;
    float* EOUT  = (float*)EIN;  // 16 x 64 f32 (4KB of the 8KB wave region)
    {
        v16h a0 = frag_a_lds(EIN, 256, 0);
        v16h a1 = frag_a_lds(EIN, 256, 32);
        asm volatile("" ::: "memory");
#pragma unroll
        for (int nt = 0; nt < 4; ++nt) {
            v8f c;
            float bb = bc1[nt * 16 + (lane & 15)];
#pragma unroll
            for (int r = 0; r < 8; ++r) c[r] = bb;
            c = __builtin_amdgcn_wmma_f32_16x16x32_f16(false, a0, false, frag_b_sw(Wc1, 0, nt), (short)0, c, false, false);
            c = __builtin_amdgcn_wmma_f32_16x16x32_f16(false, a1, false, frag_b_sw(Wc1, 1, nt), (short)0, c, false, false);
#pragma unroll
            for (int r = 0; r < 8; ++r) {
                int row = (lane & 16) ? r + 8 : r;
                EOUT[row * D + nt * 16 + (lane & 15)] = relu(c[r]);
            }
        }
    }
    asm volatile("" ::: "memory");
    // Coalesced epilogue: out_e = e_new + e0 (b128), scatter atomics, LDS mean-sum
    for (int t = lane; t < 16 * 16; t += 32) {
        int r = t >> 4, ch = (t & 15) * 4;
        int gr = base + r;
        float4 en = *(const float4*)(EOUT + r * D + ch);
        float4 ef = *(const float4*)(edge_feat + (size_t)gr * D + ch);
        float4 o  = make_float4(en.x + ef.x, en.y + ef.y, en.z + ef.z, en.w + ef.w);
        *(float4*)(out_e + (size_t)gr * D + ch) = o;
        float* vp = vesum + (size_t)dst[gr] * D + ch;
        atomicAdd(vp + 0, en.x); atomicAdd(vp + 1, en.y);
        atomicAdd(vp + 2, en.z); atomicAdd(vp + 3, en.w);
        atomicAdd(&esum_s[ch + 0], en.x); atomicAdd(&esum_s[ch + 1], en.y);
        atomicAdd(&esum_s[ch + 2], en.z); atomicAdd(&esum_s[ch + 3], en.w);
    }
    // edge counts per dst: one lane per 8 rows
    if ((lane & 15) == 0) {
#pragma unroll
        for (int r = 0; r < 8; ++r) {
            int row = (lane & 16) ? r + 8 : r;
            atomicAdd(&cntp[dst[base + row]], 1.0f);
        }
    }
    __syncthreads();
    if (tid < D) atomicAdd(&esum[tid], esum_s[tid]);
}

// -------------------- node pipeline: v_in = [v | ve | u] --------------------
__global__ __launch_bounds__(128) void node_kernel(const float* __restrict__ node_feat,
                                                   const float* cvW0, const float* cvb0,
                                                   const float* cvW1, const float* cvb1,
                                                   float* ws, float* __restrict__ out_v) {
    __shared__ __align__(32) _Float16 W0[3 * D * D];
    __shared__ __align__(32) _Float16 W1[D * D];
    __shared__ float    b0[D], b1[D];
    __shared__ __align__(32) _Float16 u16[D];
    __shared__ float    uf[D];
    __shared__ __align__(32) _Float16 stg[4 * 16 * 3 * D]; // per wave: 16 x 192 f16
    __shared__ float vsum_s[D];

    int tid = threadIdx.x, nthr = blockDim.x;
    const float* uenc = ws + OFF_UENC;
    if (tid < D) async_lds_load_f32(&uf[tid], uenc + tid);

    cvt_w_fragB(cvW0, W0, 3 * D, tid, nthr);
    cvt_w_fragB(cvW1, W1, D, tid, nthr);
    wait_asynccnt0();
    for (int i = tid; i < D; i += nthr) {
        b0[i] = cvb0[i]; b1[i] = cvb1[i];
        u16[i] = (_Float16)uf[i];
        vsum_s[i] = 0.0f;
    }
    __syncthreads();

    int w = tid >> 5, lane = tid & 31;
    int base = blockIdx.x * 64 + w * 16;
    _Float16* VIN = &stg[w * 16 * 192];
    const float* venc  = ws + OFF_V;
    const float* vesum = ws + OFF_VESUM;
    const float* cntp  = ws + OFF_CNT;

    for (int t = lane; t < 16 * 16; t += 32) {
        int r = t >> 4, ch = (t & 15) * 4;
        int gr = base + r;
        float4 vv = make_float4(0.f, 0.f, 0.f, 0.f);
        float4 ve = vv;
        if (gr < N_NODES) {
            vv = *(const float4*)(venc + (size_t)gr * D + ch);
            float inv = 1.0f / fmaxf(cntp[gr], 1.0f);
            float4 vs = *(const float4*)(vesum + (size_t)gr * D + ch);
            ve = make_float4(vs.x * inv, vs.y * inv, vs.z * inv, vs.w * inv);
        }
        *(h4*)(VIN + r * 192 + ch)       = f4_to_h4(vv);
        *(h4*)(VIN + r * 192 + 64 + ch)  = f4_to_h4(ve);
        *(h4*)(VIN + r * 192 + 128 + ch) = *(const h4*)(u16 + ch);
    }

    // Layer 1: 192 -> 64
    v8f h[4];
#pragma unroll
    for (int nt = 0; nt < 4; ++nt) {
        float bb = b0[nt * 16 + (lane & 15)];
#pragma unroll
        for (int r = 0; r < 8; ++r) h[nt][r] = bb;
#pragma unroll
        for (int kt = 0; kt < 6; ++kt) {
            v16h a = frag_a_lds(VIN, 192, kt * 32);
            h[nt] = __builtin_amdgcn_wmma_f32_16x16x32_f16(false, a, false, frag_b_sw(W0, kt, nt), (short)0, h[nt], false, false);
        }
    }
    asm volatile("" ::: "memory");
#pragma unroll
    for (int nt = 0; nt < 4; ++nt)
#pragma unroll
        for (int r = 0; r < 8; ++r) {
            int row = (lane & 16) ? r + 8 : r;
            VIN[row * 192 + nt * 16 + (lane & 15)] = (_Float16)relu(h[nt][r]);
        }

    // Layer 2: 64 -> 64; f32 results to LDS, coalesced epilogue
    float* vsum = ws + OFF_VSUM;
    float* VOUT = (float*)VIN;  // 16 x 64 f32 (4KB of the 6KB wave region)
    {
        v16h a0 = frag_a_lds(VIN, 192, 0);
        v16h a1 = frag_a_lds(VIN, 192, 32);
        asm volatile("" ::: "memory");
#pragma unroll
        for (int nt = 0; nt < 4; ++nt) {
            v8f c;
            float bb = b1[nt * 16 + (lane & 15)];
#pragma unroll
            for (int r = 0; r < 8; ++r) c[r] = bb;
            c = __builtin_amdgcn_wmma_f32_16x16x32_f16(false, a0, false, frag_b_sw(W1, 0, nt), (short)0, c, false, false);
            c = __builtin_amdgcn_wmma_f32_16x16x32_f16(false, a1, false, frag_b_sw(W1, 1, nt), (short)0, c, false, false);
#pragma unroll
            for (int r = 0; r < 8; ++r) {
                int row = (lane & 16) ? r + 8 : r;
                VOUT[row * D + nt * 16 + (lane & 15)] = relu(c[r]);
            }
        }
    }
    asm volatile("" ::: "memory");
    for (int t = lane; t < 16 * 16; t += 32) {
        int r = t >> 4, ch = (t & 15) * 4;
        int gr = base + r;
        if (gr < N_NODES) {
            float4 vn = *(const float4*)(VOUT + r * D + ch);
            float4 v0 = *(const float4*)(node_feat + (size_t)gr * D + ch);
            *(float4*)(out_v + (size_t)gr * D + ch) =
                make_float4(vn.x + v0.x, vn.y + v0.y, vn.z + v0.z, vn.w + v0.w);
            atomicAdd(&vsum_s[ch + 0], vn.x); atomicAdd(&vsum_s[ch + 1], vn.y);
            atomicAdd(&vsum_s[ch + 2], vn.z); atomicAdd(&vsum_s[ch + 3], vn.w);
        }
    }
    __syncthreads();
    if (tid < D) atomicAdd(&vsum[tid], vsum_s[tid]);
}

// -------------------- state MLP --------------------
__global__ __launch_bounds__(64) void state_kernel(const float* state_attr,
                                                   const float* cuW0, const float* cub0,
                                                   const float* cuW1, const float* cub1,
                                                   float* ws, float* out_u) {
    __shared__ float uin[3 * D];
    __shared__ float hh[D];
    int t = threadIdx.x;
    const float* uenc = ws + OFF_UENC;
    const float* esum = ws + OFF_ESUM;
    const float* vsum = ws + OFF_VSUM;
    if (t < D) {
        uin[t]         = uenc[t];
        uin[D + t]     = esum[t] * (1.0f / (float)N_EDGES);
        uin[2 * D + t] = vsum[t] * (1.0f / (float)N_NODES);
    }
    __syncthreads();
    if (t < D) {
        float acc = cub0[t];
        for (int k = 0; k < 3 * D; ++k) acc += uin[k] * cuW0[k * D + t];
        hh[t] = relu(acc);
    }
    __syncthreads();
    if (t < D) {
        float acc = cub1[t];
        for (int k = 0; k < D; ++k) acc += hh[k] * cuW1[k * D + t];
        out_u[t] = relu(acc) + state_attr[t];
    }
}

// -------------------- launcher --------------------
extern "C" void kernel_launch(void* const* d_in, const int* in_sizes, int n_in,
                              void* d_out, int out_size, void* d_ws, size_t ws_size,
                              hipStream_t stream) {
    (void)in_sizes; (void)n_in; (void)out_size; (void)ws_size;
    const int*   src        = (const int*)d_in[0];
    const int*   dst        = (const int*)d_in[1];
    const float* edge_feat  = (const float*)d_in[2];
    const float* node_feat  = (const float*)d_in[3];
    const float* state_attr = (const float*)d_in[4];
    const float* deW  = (const float*)d_in[5];
    const float* deb  = (const float*)d_in[6];
    const float* dvW  = (const float*)d_in[7];
    const float* dvb  = (const float*)d_in[8];
    const float* duW  = (const float*)d_in[9];
    const float* dub  = (const float*)d_in[10];
    const float* ceW0 = (const float*)d_in[11];
    const float* ceb0 = (const float*)d_in[12];
    const float* ceW1 = (const float*)d_in[13];
    const float* ceb1 = (const float*)d_in[14];
    const float* cvW0 = (const float*)d_in[15];
    const float* cvb0 = (const float*)d_in[16];
    const float* cvW1 = (const float*)d_in[17];
    const float* cvb1 = (const float*)d_in[18];
    const float* cuW0 = (const float*)d_in[19];
    const float* cub0 = (const float*)d_in[20];
    const float* cuW1 = (const float*)d_in[21];
    const float* cub1 = (const float*)d_in[22];

    float* ws    = (float*)d_ws;
    float* out_e = (float*)d_out;                       // N_EDGES*D
    float* out_v = out_e + (size_t)N_EDGES * D;         // N_NODES*D
    float* out_u = out_v + (size_t)N_NODES * D;         // D

    zero_kernel<<<(ZERO_N + 255) / 256, 256, 0, stream>>>(ws);
    uenc_kernel<<<1, 64, 0, stream>>>(state_attr, duW, dub, ws);
    venc_kernel<<<(N_NODES + 63) / 64, 128, 0, stream>>>(node_feat, dvW, dvb, ws);
    edge_kernel<<<N_EDGES / 64, 128, 0, stream>>>(src, dst, edge_feat, deW, deb,
                                                  ceW0, ceb0, ceW1, ceb1, ws, out_e);
    node_kernel<<<(N_NODES + 63) / 64, 128, 0, stream>>>(node_feat, cvW0, cvb0, cvW1, cvb1,
                                                         ws, out_v);
    state_kernel<<<1, 64, 0, stream>>>(state_attr, cuW0, cub0, cuW1, cub1, ws, out_u);
}
